// LSTMEncoder_10462540333383
// MI455X (gfx1250) — matrix operand order repaired and checked by reading, not compiled, not run on hardware
//
#include <hip/hip_runtime.h>

// Problem constants (from reference): B=256, T=512, I=24, H=128, 4H=512
#define TT 512
#define BB 256
#define HH 128
#define NG 512   // 4*H gates
#define II 24

typedef _Float16 half16 __attribute__((ext_vector_type(16)));
typedef _Float16 half8  __attribute__((ext_vector_type(8)));
typedef float    float8 __attribute__((ext_vector_type(8)));

union Frag16 { half16 v; half8 h[2]; };

static __device__ __forceinline__ float8 wmma_f16(half16 a, half16 b, float8 c) {
  // D = A(16x32 f16) * B(32x16 f16) + C(16x16 f32)
  return __builtin_amdgcn_wmma_f32_16x16x32_f16(
      /*neg_a=*/false, a, /*neg_b=*/false, b,
      /*c_mod=*/(short)0, c, /*reuse_a=*/false, /*reuse_b=*/false);
}

static __device__ __forceinline__ float fsig(float x) {
  return 1.0f / (1.0f + __expf(-x));
}
static __device__ __forceinline__ float ftanh_fast(float x) {
  float e = __expf(-2.0f * x);
  return (1.0f - e) / (1.0f + e);
}

// ---------------------------------------------------------------------------
// Input-projection GEMM: out[t][b][n] = sum_k in[m][k] * w[n][k] + bias[n]
// m = b*T + t, M = B*T = 131072, N = 512, K = KACT (padded to KPAD in LDS).
// Block: 128 M-rows x 64 N-cols, 8 waves (wave w -> M rows [16w,16w+16), all 64 N).
// blockIdx.z selects direction (fwd/bwd weights + output buffer).
// ---------------------------------------------------------------------------
template <int KACT, int KPAD>
__global__ __launch_bounds__(256) void gx_gemm(
    const float* __restrict__ in,                                   // [M][KACT]
    const float* __restrict__ w_f, const float* __restrict__ w_b,   // [512][KACT]
    const float* __restrict__ bias_f, const float* __restrict__ bias_b,
    float* __restrict__ out_f, float* __restrict__ out_b)           // [T][B][512]
{
  __shared__ _Float16 Alds[128 * 32];  // [128 rows][32 k]
  __shared__ _Float16 Blds[64 * 32];   // [64  n   ][32 k]

  const int tid  = threadIdx.x;
  const int wv   = tid >> 5;
  const int lane = tid & 31;
  const int r    = lane & 15;
  const int hf   = lane >> 4;
  const int Mbase = blockIdx.x * 128;
  const int Nbase = blockIdx.y * 64;
  const int dir   = blockIdx.z;

  const float* w    = dir ? w_b    : w_f;
  const float* bias = dir ? bias_b : bias_f;
  float*       out  = dir ? out_b  : out_f;

  float8 acc[4];
  #pragma unroll
  for (int nt = 0; nt < 4; ++nt)
    #pragma unroll
    for (int v = 0; v < 8; ++v) acc[nt][v] = 0.0f;

  for (int k0 = 0; k0 < KPAD; k0 += 32) {
    __syncthreads();
    {   // stage A slab [128][32] with f32->f16 convert (+zero K padding)
      const int row = tid >> 1;
      const int ks  = (tid & 1) * 16;
      const float* src = in + (size_t)(Mbase + row) * KACT;
      #pragma unroll
      for (int e = 0; e < 16; ++e) {
        const int k = k0 + ks + e;
        const float v = (k < KACT) ? src[k] : 0.0f;
        Alds[row * 32 + ks + e] = (_Float16)v;
      }
    }
    {   // stage B slab: w rows [Nbase..Nbase+64) x k-slice (already K-major)
      const int n  = tid >> 2;
      const int ks = (tid & 3) * 8;
      const float* src = w + (size_t)(Nbase + n) * KACT;
      #pragma unroll
      for (int e = 0; e < 8; ++e) {
        const int k = k0 + ks + e;
        const float v = (k < KACT) ? src[k] : 0.0f;
        Blds[n * 32 + ks + e] = (_Float16)v;
      }
    }
    __syncthreads();

    Frag16 a;   // ISA 16-bit A layout: lane<16 -> K{0..7,16..23}, lane>=16 -> K{8..15,24..31}
    a.h[0] = *(const half8*)&Alds[(wv * 16 + r) * 32 + hf * 8];
    a.h[1] = *(const half8*)&Alds[(wv * 16 + r) * 32 + 16 + hf * 8];
    #pragma unroll
    for (int nt = 0; nt < 4; ++nt) {
      Frag16 b;  // lane<16 -> K{0..7,16..23} of slice (N=r), lane>=16 -> K{8..15,24..31}
      b.h[0] = *(const half8*)&Blds[(nt * 16 + r) * 32 + hf * 8];
      b.h[1] = *(const half8*)&Blds[(nt * 16 + r) * 32 + 16 + hf * 8];
      acc[nt] = wmma_f16(a.v, b.v, acc[nt]);
    }
  }

  // Store with bias, remapping m=b*T+t -> [t][b][n]
  #pragma unroll
  for (int nt = 0; nt < 4; ++nt) {
    const int n  = Nbase + nt * 16 + r;
    const float bv = bias[n];
    #pragma unroll
    for (int v = 0; v < 8; ++v) {
      const int m = Mbase + wv * 16 + v + 8 * hf;
      const int t = m & (TT - 1);
      const int b = m >> 9;
      out[((size_t)(t * BB + b)) * NG + n] = acc[nt][v] + bv;
    }
  }
}

// ---------------------------------------------------------------------------
// Recurrent scan. One WG owns 16 batch rows for one direction, loops over T.
// w_hh (512x128) resident in LDS as f16 (128KB); h state f16 in LDS; c in regs.
// Per step: acc = gx[t] (f32), acc += h @ w_hh^T via 4x4 WMMAs, gates through
// 32KB LDS, fused sigmoid/tanh elementwise, write y / h_n / c_n.
// gx[t+1] slab is prefetched (global_prefetch_b8) during step t's WMMA chain
// to pull the serial-critical-path HBM latency off the next iteration.
// ---------------------------------------------------------------------------
__global__ __launch_bounds__(256) void lstm_scan(
    const float* __restrict__ gx_f, const float* __restrict__ gx_b,   // [T][B][512]
    const float* __restrict__ whh_f, const float* __restrict__ whh_b, // [512][128]
    float* __restrict__ y,    // [B][T][256] concat output (dir adds +128)
    float* __restrict__ hn,   // [4][B][128]
    float* __restrict__ cn,   // [4][B][128]
    int layer)
{
  extern __shared__ char smem[];
  _Float16* wlds = (_Float16*)smem;                              // 512*128 halfs
  _Float16* hlds = (_Float16*)(smem + NG * HH * 2);              // 16*128 halfs
  float*    gbuf = (float*)(smem + NG * HH * 2 + 16 * HH * 2);   // 16*512 f32

  const int tid  = threadIdx.x;
  const int wv   = tid >> 5;
  const int lane = tid & 31;
  const int r    = lane & 15;
  const int hf   = lane >> 4;
  const int bb   = blockIdx.x * 16;
  const int dir  = blockIdx.y;

  const float* gx  = dir ? gx_b  : gx_f;
  const float* whh = dir ? whh_b : whh_f;

  // Preload recurrent weights (K-major already: wlds[n][k] = w_hh[n][k])
  for (int i = tid; i < NG * HH; i += 256) wlds[i] = (_Float16)whh[i];
  for (int i = tid; i < 16 * HH; i += 256) hlds[i] = (_Float16)0.0f;

  // Elementwise ownership: thread -> (row = tid>>4, 8 consecutive j)
  const int erow = tid >> 4;
  const int ejb  = (tid & 15) * 8;
  float c[8];
  #pragma unroll
  for (int e = 0; e < 8; ++e) c[e] = 0.0f;

  // Per-thread prefetch slot: 256 threads x 128B lines cover the 32KB step slab
  const int prow = tid >> 4;          // 0..15 batch row within slice
  const int pcol = (tid & 15) * 32;   // 16 x 128B lines per row

  __syncthreads();

  for (int s = 0; s < TT; ++s) {
    const int t = dir ? (TT - 1 - s) : s;

    // ---- gate matmul: acc = gx[t] + h @ w_hh^T (wave wv -> cols [64wv,64wv+64))
    float8 acc[4];
    #pragma unroll
    for (int nt = 0; nt < 4; ++nt) {
      const int n = wv * 64 + nt * 16 + r;
      #pragma unroll
      for (int v = 0; v < 8; ++v)
        acc[nt][v] = gx[((size_t)t * BB + bb + v + 8 * hf) * NG + n];
    }

    // Prefetch next step's gx slab (uniform branch; overlaps WMMA chain below)
    if (s + 1 < TT) {
      const int tn = dir ? (TT - 2 - s) : (s + 1);
      __builtin_prefetch(&gx[((size_t)tn * BB + bb + prow) * NG + pcol], 0, 3);
    }

    #pragma unroll
    for (int k0 = 0; k0 < HH; k0 += 32) {
      Frag16 a;
      a.h[0] = *(const half8*)&hlds[r * HH + k0 + hf * 8];
      a.h[1] = *(const half8*)&hlds[r * HH + k0 + 16 + hf * 8];
      #pragma unroll
      for (int nt = 0; nt < 4; ++nt) {
        const int n = wv * 64 + nt * 16 + r;
        Frag16 b;
        b.h[0] = *(const half8*)&wlds[n * HH + k0 + hf * 8];
        b.h[1] = *(const half8*)&wlds[n * HH + k0 + 16 + hf * 8];
        acc[nt] = wmma_f16(a.v, b.v, acc[nt]);
      }
    }
    #pragma unroll
    for (int nt = 0; nt < 4; ++nt) {
      const int n = wv * 64 + nt * 16 + r;
      #pragma unroll
      for (int v = 0; v < 8; ++v)
        gbuf[(v + 8 * hf) * NG + n] = acc[nt][v];
    }
    __syncthreads();

    // ---- fused LSTM cell elementwise (PyTorch gate order i,f,g,o)
    #pragma unroll
    for (int e = 0; e < 8; ++e) {
      const int j = ejb + e;
      const float gi = gbuf[erow * NG + j];
      const float gf = gbuf[erow * NG + 128 + j];
      const float gg = gbuf[erow * NG + 256 + j];
      const float go = gbuf[erow * NG + 384 + j];
      const float iv = fsig(gi);
      const float fv = fsig(gf);
      const float gv = ftanh_fast(gg);
      const float ov = fsig(go);
      c[e] = fv * c[e] + iv * gv;
      const float h = ov * ftanh_fast(c[e]);
      hlds[erow * HH + j] = (_Float16)h;
      y[((size_t)(bb + erow) * TT + t) * 256 + dir * HH + j] = h;
      if (s == TT - 1) {
        hn[((size_t)(layer * 2 + dir) * BB + bb + erow) * HH + j] = h;
        cn[((size_t)(layer * 2 + dir) * BB + bb + erow) * HH + j] = c[e];
      }
    }
    __syncthreads();  // protect hlds (next matmul) and gbuf (next write)
  }
}

extern "C" void kernel_launch(void* const* d_in, const int* in_sizes, int n_in,
                              void* d_out, int out_size, void* d_ws, size_t ws_size,
                              hipStream_t stream) {
  (void)in_sizes; (void)n_in; (void)out_size; (void)ws_size;

  const float* x         = (const float*)d_in[0];
  const float* w_ih_l0f  = (const float*)d_in[1];
  const float* w_hh_l0f  = (const float*)d_in[2];
  const float* b_l0f     = (const float*)d_in[3];
  const float* w_ih_l0b  = (const float*)d_in[4];
  const float* w_hh_l0b  = (const float*)d_in[5];
  const float* b_l0b     = (const float*)d_in[6];
  const float* w_ih_l1f  = (const float*)d_in[7];
  const float* w_hh_l1f  = (const float*)d_in[8];
  const float* b_l1f     = (const float*)d_in[9];
  const float* w_ih_l1b  = (const float*)d_in[10];
  const float* w_hh_l1b  = (const float*)d_in[11];
  const float* b_l1b     = (const float*)d_in[12];

  // Workspace: gxF | gxB (each T*B*512 f32) | y0 (B*T*256 f32)
  float* gxF = (float*)d_ws;
  float* gxB = gxF + (size_t)TT * BB * NG;
  float* y0  = gxB + (size_t)TT * BB * NG;

  // Output: y1 [B][T][256] | h_n [4][B][128] | c_n [4][B][128]
  float* y1 = (float*)d_out;
  float* hn = y1 + (size_t)BB * TT * 256;
  float* cn = hn + (size_t)4 * BB * HH;

  const dim3 ggrid((BB * TT) / 128, NG / 64, 2);
  const dim3 sgrid(BB / 16, 2);
  const size_t smem = (size_t)NG * HH * 2 + 16 * HH * 2 + 16 * NG * 4;  // 164 KB

  // Layer 0
  gx_gemm<II, 32><<<ggrid, 256, 0, stream>>>(
      x, w_ih_l0f, w_ih_l0b, b_l0f, b_l0b, gxF, gxB);
  lstm_scan<<<sgrid, 256, smem, stream>>>(
      gxF, gxB, w_hh_l0f, w_hh_l0b, y0, hn, cn, 0);

  // Layer 1 (input = y0, K = 2H = 256)
  gx_gemm<2 * HH, 2 * HH><<<ggrid, 256, 0, stream>>>(
      y0, w_ih_l1f, w_ih_l1b, b_l1f, b_l1b, gxF, gxB);
  lstm_scan<<<sgrid, 256, smem, stream>>>(
      gxF, gxB, w_hh_l1f, w_hh_l1b, y1, hn, cn, 1);
}
